// ProbAttention_85091892068955
// MI455X (gfx1250) — matrix operand order repaired and checked by reading, not compiled, not run on hardware
//
#include <hip/hip_runtime.h>
#include <hip/hip_bf16.h>

typedef __attribute__((ext_vector_type(16))) _Float16 v16h;
typedef __attribute__((ext_vector_type(8)))  float    v8f;

#define NB 4
#define NH 8
#define NL 2048
#define ND 64
#define NSK 40        // sample_k = 5*ceil(ln(2048)) = 40
#define NU 40         // u = min(40, 2048)
#define NSPLIT 8      // flash split-K over L (256 keys each)
#define NSEG 8        // cumsum segments of 256 rows
#define NEG_INF (-__builtin_inff())

// ---------------------------------------------------------------------------
// Kernel 1: sampled sparsity measure M[q] = max_s(q·K[idx_s]) - sum_s(...)/L
// One wave32 per query; lane handles 2 of the 64 dims; shfl-xor reduction.
// ---------------------------------------------------------------------------
__global__ __launch_bounds__(128) void k_sample_scores(
    const float* __restrict__ Q, const float* __restrict__ K,
    const int* __restrict__ idx, float* __restrict__ M) {
  const int wave = blockIdx.x * 4 + (threadIdx.x >> 5);   // global query id
  const int lane = threadIdx.x & 31;
  if (wave >= NB * NH * NL) return;
  const int l  = wave % NL;
  const int bh = wave / NL;
  const float2 qv = ((const float2*)(Q + (size_t)wave * ND))[lane];
  const float* Kbh = K + (size_t)bh * NL * ND;
  float mx = NEG_INF, sm = 0.f;
  for (int s = 0; s < NSK; ++s) {
    const int ki = idx[l * NSK + s];
    const float2 kv = ((const float2*)(Kbh + (size_t)ki * ND))[lane];
    float p = qv.x * kv.x + qv.y * kv.y;
    p += __shfl_xor(p, 16); p += __shfl_xor(p, 8);
    p += __shfl_xor(p, 4);  p += __shfl_xor(p, 2);  p += __shfl_xor(p, 1);
    mx = fmaxf(mx, p);
    sm += p;
  }
  if (lane == 0) M[wave] = mx - sm * (1.0f / (float)NL);
}

// ---------------------------------------------------------------------------
// Kernel 2: top-40 indices of M per (b,h). LDS argmax selection loop.
// ---------------------------------------------------------------------------
__global__ __launch_bounds__(256) void k_topk(
    const float* __restrict__ M, int* __restrict__ Mtop) {
  const int bh = blockIdx.x;
  const float* m = M + (size_t)bh * NL;
  __shared__ float sv[NL];
  __shared__ float bval[256];
  __shared__ int   bidx[256];
  for (int i = threadIdx.x; i < NL; i += 256) sv[i] = m[i];
  __syncthreads();
  for (int t = 0; t < NU; ++t) {
    float best = NEG_INF; int bi = NL;
    for (int i = threadIdx.x; i < NL; i += 256) {
      const float v = sv[i];
      if (v > best || (v == best && i < bi)) { best = v; bi = i; }
    }
    bval[threadIdx.x] = best; bidx[threadIdx.x] = bi;
    __syncthreads();
    for (int s = 128; s > 0; s >>= 1) {
      if (threadIdx.x < s) {
        const float ov = bval[threadIdx.x + s]; const int oi = bidx[threadIdx.x + s];
        if (ov > bval[threadIdx.x] || (ov == bval[threadIdx.x] && oi < bidx[threadIdx.x])) {
          bval[threadIdx.x] = ov; bidx[threadIdx.x] = oi;
        }
      }
      __syncthreads();
    }
    if (threadIdx.x == 0) { Mtop[bh * NU + t] = bidx[0]; sv[bidx[0]] = NEG_INF; }
    __syncthreads();
  }
}

// ---------------------------------------------------------------------------
// Hierarchical cumsum of V along L, written transposed to out (B,L,H,D).
// 3a: per (bh, segment of 256 rows): 4 sub-chunk (64-row) sums + segment total
// 3b: per bh: exclusive prefix over the 8 segment totals
// 3c: per (bh, segment): emit running sums with sub-chunk bases
// ---------------------------------------------------------------------------
__global__ __launch_bounds__(256) void k_scan_part(
    const float* __restrict__ V, float* __restrict__ Sub, float* __restrict__ Seg) {
  const int bh = blockIdx.x, s = blockIdx.y;
  const int d = threadIdx.x & 63;
  const int c = threadIdx.x >> 6;                  // sub-chunk 0..3 (64 rows)
  const float* vp = V + (size_t)bh * NL * ND + (size_t)(s * 256 + c * 64) * ND + d;
  float sum = 0.f;
  for (int i = 0; i < 64; ++i) sum += vp[(size_t)i * ND];
  Sub[((size_t)(bh * NSEG + s) * 4 + c) * ND + d] = sum;
  __shared__ float cs[4][ND];
  cs[c][d] = sum;
  __syncthreads();
  if (c == 0)
    Seg[(size_t)(bh * NSEG + s) * ND + d] = cs[0][d] + cs[1][d] + cs[2][d] + cs[3][d];
}

__global__ __launch_bounds__(64) void k_scan_pre(
    const float* __restrict__ Seg, float* __restrict__ SegPre) {
  const int bh = blockIdx.x;
  const int d = threadIdx.x;
  float run = 0.f;
  for (int s = 0; s < NSEG; ++s) {
    const size_t o = (size_t)(bh * NSEG + s) * ND + d;
    SegPre[o] = run;
    run += Seg[o];
  }
}

__global__ __launch_bounds__(256) void k_scan_emit(
    const float* __restrict__ V, const float* __restrict__ Sub,
    const float* __restrict__ SegPre, float* __restrict__ out) {
  const int bh = blockIdx.x, s = blockIdx.y;
  const int b = bh / NH, h = bh % NH;
  const int d = threadIdx.x & 63;
  const int c = threadIdx.x >> 6;
  float run = SegPre[(size_t)(bh * NSEG + s) * ND + d];
  for (int cc = 0; cc < c; ++cc)
    run += Sub[((size_t)(bh * NSEG + s) * 4 + cc) * ND + d];
  const int l0 = s * 256 + c * 64;
  const float* vp = V + (size_t)bh * NL * ND + (size_t)l0 * ND + d;
  for (int i = 0; i < 64; ++i) {
    run += vp[(size_t)i * ND];
    out[(((size_t)b * NL + (l0 + i)) * NH + h) * ND + d] = run;
  }
}

// ---------------------------------------------------------------------------
// Kernel 4a: split-K flash attention partials with WMMA f16->f32.
// Grid (B*H, 3 m-tiles, NSPLIT): one wave32 per block handles 256 keys.
// Emits unnormalized partial acc + per-row (max, sum) to workspace.
// WMMA operand layouts per CDNA5 ISA 7.12.2 (wave32).
// ---------------------------------------------------------------------------
__global__ __launch_bounds__(32) void k_attn_part(
    const float* __restrict__ Q, const float* __restrict__ K,
    const float* __restrict__ V, const int* __restrict__ Mtop,
    float* __restrict__ Pacc, float* __restrict__ Pmax, float* __restrict__ Psum) {
  const int bh    = blockIdx.x;
  const int mtile = blockIdx.y;              // 0..2 (rows 0..47, valid 0..39)
  const int split = blockIdx.z;              // key range [split*256, +256)
  const int lane = threadIdx.x;
  const int lo   = lane & 15;
  const int hi   = lane >> 4;                // 0 or 1

  const float* Qbh = Q + (size_t)bh * NL * ND;
  const float* Kbh = K + (size_t)bh * NL * ND;
  const float* Vbh = V + (size_t)bh * NL * ND;
  const int*   mt  = Mtop + bh * NU;

  // ---- A operand rows: tile-local row = lane&15 ----
  const int arow = mtile * 16 + lo;
  const int aq   = (arow < NU) ? mt[arow] : -1;

  // Q tile in A layout, two K-chunks of 32 over D=64.
  // half h: k = (h>>3)*16 + hi*8 + (h&7)
  v16h qa[2];
#pragma unroll
  for (int kc = 0; kc < 2; ++kc) {
    if (aq >= 0) {
      const float* qp = Qbh + (size_t)aq * ND + kc * 32 + hi * 8;
#pragma unroll
      for (int t = 0; t < 8; ++t) qa[kc][t]     = (_Float16)qp[t];
#pragma unroll
      for (int t = 0; t < 8; ++t) qa[kc][8 + t] = (_Float16)qp[16 + t];
    } else {
#pragma unroll
      for (int t = 0; t < 16; ++t) qa[kc][t] = (_Float16)0.f;
    }
  }

  // ---- C/D layout rows this lane owns: m = hi*8 + i ----
  int crow_q[8];
#pragma unroll
  for (int i = 0; i < 8; ++i) {
    const int ur = mtile * 16 + hi * 8 + i;
    crow_q[i] = (ur < NU) ? mt[ur] : -1;
  }

  // causal early exit: max selected query index in this tile (wave-uniform)
  int tq = -1;
#pragma unroll
  for (int i = 0; i < 8; ++i) tq = max(tq, crow_q[i]);
  tq = max(tq, __shfl_xor(tq, 1)); tq = max(tq, __shfl_xor(tq, 2));
  tq = max(tq, __shfl_xor(tq, 4)); tq = max(tq, __shfl_xor(tq, 8));
  tq = max(tq, __shfl_xor(tq, 16));

  float rmax[8], rsum[8];
#pragma unroll
  for (int i = 0; i < 8; ++i) { rmax[i] = NEG_INF; rsum[i] = 0.f; }

  v8f acc[4] = {};                            // 4 d-tiles of 16

  __shared__ _Float16 lp[16 * 32];            // P tile: 16 rows x 32 keys

  const int s0   = split * (NL / NSPLIT);
  const int kend = min(s0 + NL / NSPLIT, tq + 1);   // chunk needed iff kb <= tq

  for (int kb = s0; kb < kend; kb += 32) {
    if (kb + 32 < kend) {                     // prefetch next chunk
      __builtin_prefetch(Kbh + (size_t)(kb + 32) * ND, 0, 1);
      __builtin_prefetch(Vbh + (size_t)(kb + 32) * ND, 0, 1);
    }
    // ---- scores: S[nt] (16x16), A=Q tile, B=K^T tile ----
    v8f S[2];
#pragma unroll
    for (int nt = 0; nt < 2; ++nt) {
      v8f c = {};
#pragma unroll
      for (int kc = 0; kc < 2; ++kc) {
        // B: n = lane&15 -> key; half j -> d = kc*32 + (j>>3)*16 + hi*8 + (j&7)
        v16h bt;
        const float* kp = Kbh + (size_t)(kb + nt * 16 + lo) * ND + kc * 32 + hi * 8;
#pragma unroll
        for (int j = 0; j < 16; ++j) bt[j] = (_Float16)kp[j < 8 ? j : j + 8];
        c = __builtin_amdgcn_wmma_f32_16x16x32_f16(false, qa[kc], false, bt,
                                                   (short)0, c, false, false);
      }
      S[nt] = c;
    }

    // ---- causal mask + online softmax (per row m = hi*8 + i) ----
    float rowscale[8];
#pragma unroll
    for (int i = 0; i < 8; ++i) {
      const int qi = crow_q[i];
      float s0v = S[0][i], s1v = S[1][i];
      if (qi < 0 || kb + lo      > qi) s0v = NEG_INF;
      if (qi < 0 || kb + 16 + lo > qi) s1v = NEG_INF;
      float mx = fmaxf(s0v, s1v);
      mx = fmaxf(mx, __shfl_xor(mx, 1)); mx = fmaxf(mx, __shfl_xor(mx, 2));
      mx = fmaxf(mx, __shfl_xor(mx, 4)); mx = fmaxf(mx, __shfl_xor(mx, 8));
      const float nm = fmaxf(rmax[i], mx);
      const bool live = nm > NEG_INF;
      const float scale = live ? __expf(rmax[i] - nm) : 1.0f;
      const float p0 = live ? __expf(s0v - nm) : 0.f;
      const float p1 = live ? __expf(s1v - nm) : 0.f;
      float ps = p0 + p1;
      ps += __shfl_xor(ps, 1); ps += __shfl_xor(ps, 2);
      ps += __shfl_xor(ps, 4); ps += __shfl_xor(ps, 8);
      rsum[i] = rsum[i] * scale + ps;
      rmax[i] = nm;
      rowscale[i] = scale;
      const int m = hi * 8 + i;
      lp[m * 32 + lo]      = (_Float16)p0;
      lp[m * 32 + 16 + lo] = (_Float16)p1;
    }
    // rescale accumulators for new row max
#pragma unroll
    for (int dt = 0; dt < 4; ++dt)
#pragma unroll
      for (int i = 0; i < 8; ++i) acc[dt][i] *= rowscale[i];

    __syncthreads();   // LDS writes visible (single wave: forces dscnt wait)

    // ---- re-read P in A layout: row = lane&15; k = (h>>3)*16 + hi*8 + (h&7)
    v16h pa;
    {
      const int k0 = hi * 8;
#pragma unroll
      for (int t = 0; t < 8; ++t) pa[t]     = lp[lo * 32 + k0 + t];
#pragma unroll
      for (int t = 0; t < 8; ++t) pa[8 + t] = lp[lo * 32 + 16 + k0 + t];
    }
    __syncthreads();

    // ---- acc += P (16x32) x V chunk (32 x 64), 4 d-tiles ----
#pragma unroll
    for (int dt = 0; dt < 4; ++dt) {
      v16h vt;  // B: n = lane&15 -> d col; half j -> key = kb + (j>>3)*16 + hi*8 + (j&7)
      const float* vp = Vbh + (size_t)(kb + hi * 8) * ND + dt * 16 + lo;
#pragma unroll
      for (int j = 0; j < 16; ++j) vt[j] = (_Float16)vp[(size_t)(j < 8 ? j : j + 8) * ND];
      acc[dt] = __builtin_amdgcn_wmma_f32_16x16x32_f16(false, pa, false, vt,
                                                       (short)0, acc[dt], false, false);
    }
  }

  // ---- write partials (unnormalized) ----
  const size_t pb = (size_t)(bh * 3 + mtile) * NSPLIT + split;
#pragma unroll
  for (int i = 0; i < 8; ++i) {
    const int r = hi * 8 + i;
#pragma unroll
    for (int dt = 0; dt < 4; ++dt)
      Pacc[(pb * 16 + r) * ND + dt * 16 + lo] = acc[dt][i];
    if (lo == 0) {
      Pmax[pb * 16 + r] = rmax[i];
      Psum[pb * 16 + r] = rsum[i];
    }
  }
}

// ---------------------------------------------------------------------------
// Kernel 4b: merge NSPLIT partials per (bh, m-tile), normalize, scatter rows.
// ---------------------------------------------------------------------------
__global__ __launch_bounds__(64) void k_attn_combine(
    const float* __restrict__ Pacc, const float* __restrict__ Pmax,
    const float* __restrict__ Psum, const int* __restrict__ Mtop,
    float* __restrict__ out) {
  const int bm = blockIdx.x;                 // bh*3 + mtile
  const int bh = bm / 3, mtile = bm % 3;
  const int b = bh / NH, h = bh % NH;
  const int d = threadIdx.x;                 // 0..63

  float gmax[16];
#pragma unroll
  for (int r = 0; r < 16; ++r) gmax[r] = NEG_INF;
  for (int s = 0; s < NSPLIT; ++s) {
    const size_t pb = (size_t)bm * NSPLIT + s;
#pragma unroll
    for (int r = 0; r < 16; ++r) gmax[r] = fmaxf(gmax[r], Pmax[pb * 16 + r]);
  }

  float gsum[16], accd[16];
#pragma unroll
  for (int r = 0; r < 16; ++r) { gsum[r] = 0.f; accd[r] = 0.f; }

  for (int s = 0; s < NSPLIT; ++s) {
    const size_t pb = (size_t)bm * NSPLIT + s;
#pragma unroll
    for (int r = 0; r < 16; ++r) {
      const int ur = mtile * 16 + r;
      if (ur >= NU) continue;                       // padding row
      const float pm = Pmax[pb * 16 + r];
      const float w = (pm > NEG_INF) ? __expf(pm - gmax[r]) : 0.f;
      gsum[r] += w * Psum[pb * 16 + r];
      accd[r] += w * Pacc[(pb * 16 + r) * ND + d];
    }
  }

#pragma unroll
  for (int r = 0; r < 16; ++r) {
    const int ur = mtile * 16 + r;
    if (ur < NU) {
      const int qi = Mtop[bh * NU + ur];
      out[(((size_t)b * NL + qi) * NH + h) * ND + d] = accd[r] / gsum[r];
    }
  }
}

// ---------------------------------------------------------------------------
extern "C" void kernel_launch(void* const* d_in, const int* in_sizes, int n_in,
                              void* d_out, int out_size, void* d_ws, size_t ws_size,
                              hipStream_t stream) {
  const float* Q   = (const float*)d_in[0];
  const float* K   = (const float*)d_in[1];
  const float* V   = (const float*)d_in[2];
  const int*   idx = (const int*)d_in[3];
  float* out = (float*)d_out;

  // workspace carve-up (256B aligned)
  char* wp = (char*)d_ws;
  size_t o = 0;
  auto carve = [&](size_t bytes) { size_t r = o; o += (bytes + 255) & ~(size_t)255; return r; };
  float* Mbuf   = (float*)(wp + carve((size_t)NB * NH * NL * sizeof(float)));
  int*   Mtop   = (int*)  (wp + carve((size_t)NB * NH * NU * sizeof(int)));
  float* Pacc   = (float*)(wp + carve((size_t)NB * NH * 3 * NSPLIT * 16 * ND * sizeof(float)));
  float* Pmax   = (float*)(wp + carve((size_t)NB * NH * 3 * NSPLIT * 16 * sizeof(float)));
  float* Psum   = (float*)(wp + carve((size_t)NB * NH * 3 * NSPLIT * 16 * sizeof(float)));
  float* Sub    = (float*)(wp + carve((size_t)NB * NH * NSEG * 4 * ND * sizeof(float)));
  float* Seg    = (float*)(wp + carve((size_t)NB * NH * NSEG * ND * sizeof(float)));
  float* SegPre = (float*)(wp + carve((size_t)NB * NH * NSEG * ND * sizeof(float)));

  k_sample_scores<<<(NB * NH * NL) / 4, 128, 0, stream>>>(Q, K, idx, Mbuf);
  k_topk<<<NB * NH, 256, 0, stream>>>(Mbuf, Mtop);

  k_scan_part<<<dim3(NB * NH, NSEG), 256, 0, stream>>>(V, Sub, Seg);
  k_scan_pre<<<NB * NH, 64, 0, stream>>>(Seg, SegPre);
  k_scan_emit<<<dim3(NB * NH, NSEG), 256, 0, stream>>>(V, Sub, SegPre, out);

  k_attn_part<<<dim3(NB * NH, 3, NSPLIT), 32, 0, stream>>>(Q, K, V, Mtop, Pacc, Pmax, Psum);
  k_attn_combine<<<NB * NH * 3, 64, 0, stream>>>(Pacc, Pmax, Psum, Mtop, out);
}
